// ELSA_40630390620857
// MI455X (gfx1250) — matrix operand order repaired and checked by reading, not compiled, not run on hardware
//
#include <hip/hip_runtime.h>
#include <hip/hip_bf16.h>

// ---------------- problem constants ----------------
#define Bsz   16
#define Hh    56
#define Ww    56
#define DIM   384
#define G     12
#define KS    5
#define DIMQK 256
#define DIMV  384
#define GW    8
#define CG    32
#define PIX   (Bsz * Hh * Ww)          // 50176
#define NQKV  (2 * DIMQK + DIMV)       // 896
#define SCALE 0.17677669529663687f     // 32^-0.5

// ---------------- WMMA types ----------------
typedef __attribute__((ext_vector_type(16))) __bf16 v16bf;
typedef __attribute__((ext_vector_type(8)))  float  v8f;

union FragBF { v16bf v; uint4 q[2]; };

__device__ __forceinline__ unsigned short f2bf(float f) {
  unsigned u = __float_as_uint(f);
  u += 0x7FFFu + ((u >> 16) & 1u);     // round-to-nearest-even
  return (unsigned short)(u >> 16);
}
__device__ __forceinline__ unsigned pack_bf16x2(float a, float b) {
  return ((unsigned)f2bf(b) << 16) | (unsigned)f2bf(a);
}

// ---------------- fp32 -> bf16 conversion (zero-pads dst to npad) ----------------
__global__ __launch_bounds__(256) void f32_to_bf16_pad(
    const float* __restrict__ src, unsigned short* __restrict__ dst, int n, int npad)
{
  int i4 = (blockIdx.x * 256 + threadIdx.x) << 2;
  if (i4 >= npad) return;
  float4 f = make_float4(0.f, 0.f, 0.f, 0.f);
  if (i4 + 4 <= n) {
    f = *(const float4*)(src + i4);
  } else if (i4 < n) {
    f.x = src[i4];
    if (i4 + 1 < n) f.y = src[i4 + 1];
    if (i4 + 2 < n) f.z = src[i4 + 2];
  }
  uint2 p;
  p.x = pack_bf16x2(f.x, f.y);
  p.y = pack_bf16x2(f.z, f.w);
  *(uint2*)(dst + i4) = p;
}

// ---------------- bf16 WMMA GEMM: C[M,N] = A[M,K] * Bw[N,K]^T (+bias) ----------------
// A, Bw are bf16 in global memory. LDS staged via global_load_async_to_lds_b128,
// double-buffered, synchronized with s_wait_asynccnt + workgroup barrier.
#define BM   128
#define BN   128
#define BK   64
#define LSTR 72                 // LDS row stride (bf16 elems) = 144B, 16B-aligned
#define LTILE (128 * LSTR)      // elems per tile buffer

__global__ __launch_bounds__(256) void gemm_bf16_wmma(
    const unsigned short* __restrict__ A, const unsigned short* __restrict__ Bw,
    const float* __restrict__ bias, float* __restrict__ C,
    int M, int N, int K)
{
  (void)M;
  __shared__ __align__(16) unsigned short Al[2][LTILE];
  __shared__ __align__(16) unsigned short Bl[2][LTILE];

  const int tid  = threadIdx.x;
  const int lane = tid & 31;
  const int wv   = tid >> 5;
  const int wm   = wv & 3;      // 4 waves along M (32 rows each)
  const int wn   = wv >> 2;     // 2 waves along N (64 cols each)
  const int lr   = lane & 15;
  const int lh   = lane >> 4;
  const int bm   = blockIdx.x * BM;
  const int bn   = blockIdx.y * BN;

  v8f acc[2][4];
  #pragma unroll
  for (int i = 0; i < 2; ++i)
    #pragma unroll
    for (int j = 0; j < 4; ++j)
      #pragma unroll
      for (int r = 0; r < 8; ++r) acc[i][j][r] = 0.0f;

  const unsigned aLds0 = (unsigned)(size_t)(&Al[0][0]);
  const unsigned bLds0 = (unsigned)(size_t)(&Bl[0][0]);

  // async-copy one 128x64 bf16 tile of A and of B into LDS buffer `buf`
  auto stage = [&](int k0, int buf) {
    #pragma unroll
    for (int i = 0; i < 4; ++i) {
      int idx = i * 256 + tid;            // 0..1023 : 16B chunks
      int r = idx >> 3;
      int c = (idx & 7) << 3;             // bf16 elems
      unsigned lds = aLds0 + (unsigned)(buf * LTILE + r * LSTR + c) * 2u;
      const unsigned short* gp = A + (size_t)(bm + r) * K + k0 + c;
      asm volatile("global_load_async_to_lds_b128 %0, %1, off"
                   :: "v"(lds), "v"(gp) : "memory");
    }
    #pragma unroll
    for (int i = 0; i < 4; ++i) {
      int idx = i * 256 + tid;
      int r = idx >> 3;
      int c = (idx & 7) << 3;
      unsigned lds = bLds0 + (unsigned)(buf * LTILE + r * LSTR + c) * 2u;
      const unsigned short* gp = Bw + (size_t)(bn + r) * K + k0 + c;
      asm volatile("global_load_async_to_lds_b128 %0, %1, off"
                   :: "v"(lds), "v"(gp) : "memory");
    }
  };

  auto compute = [&](int buf) {
    #pragma unroll
    for (int kk = 0; kk < BK; kk += 32) {
      FragBF a[2], b[4];
      // A frag (16x32): lanes 0-15 -> K kk+[0..7]/[16..23]; lanes 16-31 -> +8
      #pragma unroll
      for (int s = 0; s < 2; ++s) {
        const unsigned short* ap =
            &Al[buf][(wm * 32 + s * 16 + lr) * LSTR + kk + lh * 8];
        a[s].q[0] = *(const uint4*)ap;
        a[s].q[1] = *(const uint4*)(ap + 16);
      }
      // B frag (32x16): lane&15 = column; lanes 0-15 -> K kk+[0..7]/[16..23]; lanes 16-31 -> +8
      #pragma unroll
      for (int j = 0; j < 4; ++j) {
        const unsigned short* bp =
            &Bl[buf][(wn * 64 + j * 16 + lr) * LSTR + kk + lh * 8];
        b[j].q[0] = *(const uint4*)bp;
        b[j].q[1] = *(const uint4*)(bp + 16);
      }
      #pragma unroll
      for (int i = 0; i < 2; ++i)
        #pragma unroll
        for (int j = 0; j < 4; ++j)
          acc[i][j] = __builtin_amdgcn_wmma_f32_16x16x32_bf16(
              false, a[i].v, false, b[j].v, (short)0, acc[i][j], false, false);
    }
  };

  stage(0, 0);
  asm volatile("s_wait_asynccnt 0x0" ::: "memory");
  __syncthreads();

  const int nk = K / BK;
  for (int t = 0; t < nk; ++t) {
    int buf = t & 1;
    if (t + 1 < nk) stage((t + 1) * BK, buf ^ 1);   // prefetch next tile (async)
    compute(buf);                                   // 16 WMMAs on current tile
    asm volatile("s_wait_asynccnt 0x0" ::: "memory");
    __syncthreads();
  }

  // epilogue: C VGPR r -> (M=r, N=lane&15) lanes 0-15 ; (M=r+8) lanes 16-31
  #pragma unroll
  for (int j = 0; j < 4; ++j) {
    int col = bn + wn * 64 + j * 16 + lr;
    if (col >= N) continue;
    float bv = bias ? bias[col] : 0.0f;
    #pragma unroll
    for (int i = 0; i < 2; ++i) {
      #pragma unroll
      for (int r = 0; r < 8; ++r) {
        int row = bm + wm * 32 + i * 16 + lh * 8 + r;
        C[(size_t)row * N + col] = acc[i][j][r] + bv;
      }
    }
  }
}

// ---------------- had = q * k * scale (fp32) ----------------
__global__ __launch_bounds__(256) void had_kernel(
    const float* __restrict__ qkv, float* __restrict__ had)
{
  int i = blockIdx.x * 256 + threadIdx.x;
  int m = i >> 6;
  int c = (i & 63) << 2;
  const float* q = qkv + (size_t)m * NQKV + c;
  float4 qv = *(const float4*)q;
  float4 kv = *(const float4*)(q + DIMQK);
  float4 o;
  o.x = qv.x * kv.x * SCALE;
  o.y = qv.y * kv.y * SCALE;
  o.z = qv.z * kv.z * SCALE;
  o.w = qv.w * kv.w * SCALE;
  *(float4*)(had + (size_t)m * DIMQK + c) = o;
}

// ---------------- grouped 5x5 conv (8->8 per group) + erf-GELU -> bf16 ----------------
__global__ __launch_bounds__(256) void conv_gelu_kernel(
    const float* __restrict__ had, const float* __restrict__ w1,
    const float* __restrict__ b1, unsigned short* __restrict__ out)
{
  const int g = blockIdx.y;                      // 32 groups
  __shared__ float wl[GW * GW * 25];
  __shared__ float bl[GW];
  for (int i = threadIdx.x; i < GW * GW * 25; i += 256)
    wl[i] = w1[(size_t)g * (GW * GW * 25) + i];
  if (threadIdx.x < GW) bl[threadIdx.x] = b1[g * GW + threadIdx.x];
  __syncthreads();

  int p = blockIdx.x * 256 + threadIdx.x;
  if (p >= PIX) return;
  int b   = p / (Hh * Ww);
  int rem = p - b * (Hh * Ww);
  int y = rem / Ww, x = rem - (rem / Ww) * Ww;

  float acc[GW];
  #pragma unroll
  for (int oc = 0; oc < GW; ++oc) acc[oc] = bl[oc];

  #pragma unroll
  for (int dy = 0; dy < KS; ++dy) {
    int yy = y + dy - 2;
    if (yy < 0 || yy >= Hh) continue;
    #pragma unroll
    for (int dx = 0; dx < KS; ++dx) {
      int xx = x + dx - 2;
      if (xx < 0 || xx >= Ww) continue;
      const float* vin = had + ((size_t)(b * Hh * Ww + yy * Ww + xx)) * DIMQK + g * GW;
      float4 f0 = *(const float4*)(vin);
      float4 f1 = *(const float4*)(vin + 4);
      float in[GW] = {f0.x, f0.y, f0.z, f0.w, f1.x, f1.y, f1.z, f1.w};
      int tap = dy * KS + dx;
      #pragma unroll
      for (int oc = 0; oc < GW; ++oc)
        #pragma unroll
        for (int ic = 0; ic < GW; ++ic)
          acc[oc] = fmaf(in[ic], wl[oc * 200 + ic * 25 + tap], acc[oc]);
    }
  }
  float gv[GW];
  #pragma unroll
  for (int oc = 0; oc < GW; ++oc) {
    float v = acc[oc];
    gv[oc] = 0.5f * v * (1.0f + erff(v * 0.70710678118654752f));   // exact GELU
  }
  uint4 o;
  o.x = pack_bf16x2(gv[0], gv[1]);
  o.y = pack_bf16x2(gv[2], gv[3]);
  o.z = pack_bf16x2(gv[4], gv[5]);
  o.w = pack_bf16x2(gv[6], gv[7]);
  *(uint4*)(out + (size_t)p * DIMQK + g * GW) = o;   // 16B aligned
}

// ---------------- softmax over 25 window positions, per head, in place ----------------
__global__ __launch_bounds__(256) void softmax25_kernel(float* __restrict__ h)
{
  int idx = blockIdx.x * 256 + threadIdx.x;
  if (idx >= PIX * G) return;
  int m = idx / G, g = idx - (idx / G) * G;
  float* row = h + (size_t)m * (G * 25) + g * 25;
  float v[25], mx = -3.4e38f;
  #pragma unroll
  for (int k = 0; k < 25; ++k) { v[k] = row[k]; mx = fmaxf(mx, v[k]); }
  float s = 0.f;
  #pragma unroll
  for (int k = 0; k < 25; ++k) { v[k] = expf(v[k] - mx); s += v[k]; }
  float inv = 1.0f / s;
  #pragma unroll
  for (int k = 0; k < 25; ++k) row[k] = v[k] * inv;
}

// ---------------- ELSA aggregation -> bf16 ----------------
__global__ __launch_bounds__(256) void elsa_kernel(
    const float* __restrict__ qkv, const float* __restrict__ attn,
    const float* __restrict__ ghost, unsigned short* __restrict__ out)
{
  const int g = blockIdx.y;                      // 12 heads
  __shared__ float gml[CG * 25];
  __shared__ float gal[CG * 25];
  for (int i = threadIdx.x; i < CG * 25; i += 256) {
    gml[i] = ghost[(size_t)(g * CG) * 25 + i];
    gal[i] = ghost[(size_t)(DIMV * 25) + (size_t)(g * CG) * 25 + i];
  }
  __syncthreads();

  int p = blockIdx.x * 256 + threadIdx.x;
  if (p >= PIX) return;
  int b   = p / (Hh * Ww);
  int rem = p - b * (Hh * Ww);
  int y = rem / Ww, x = rem - (rem / Ww) * Ww;

  float att[25];
  const float* arow = attn + (size_t)p * (G * 25) + g * 25;
  #pragma unroll
  for (int k = 0; k < 25; ++k) att[k] = arow[k];

  float acc[CG];
  #pragma unroll
  for (int c = 0; c < CG; ++c) acc[c] = 0.f;

  for (int k = 0; k < 25; ++k) {
    int yy = y + k / 5 - 2;
    int xx = x + (k % 5) - 2;
    if (yy < 0 || yy >= Hh || xx < 0 || xx >= Ww) continue;
    const float* vv = qkv + ((size_t)(b * Hh * Ww + yy * Ww + xx)) * NQKV + 2 * DIMQK + g * CG;
    float ak = att[k];
    #pragma unroll
    for (int c = 0; c < CG; ++c) {
      float filt = fmaf(gml[c * 25 + k], ak, gal[c * 25 + k]);
      acc[c] = fmaf(vv[c], filt, acc[c]);
    }
  }
  unsigned short* op = out + (size_t)p * DIMV + g * CG;   // 16B aligned
  #pragma unroll
  for (int cc = 0; cc < 4; ++cc) {
    uint4 o;
    o.x = pack_bf16x2(acc[cc * 8 + 0], acc[cc * 8 + 1]);
    o.y = pack_bf16x2(acc[cc * 8 + 2], acc[cc * 8 + 3]);
    o.z = pack_bf16x2(acc[cc * 8 + 4], acc[cc * 8 + 5]);
    o.w = pack_bf16x2(acc[cc * 8 + 6], acc[cc * 8 + 7]);
    ((uint4*)op)[cc] = o;
  }
}

// ---------------- launch ----------------
extern "C" void kernel_launch(void* const* d_in, const int* in_sizes, int n_in,
                              void* d_out, int out_size, void* d_ws, size_t ws_size,
                              hipStream_t stream) {
  (void)in_sizes; (void)n_in; (void)out_size; (void)ws_size;
  const float* x      = (const float*)d_in[0];
  const float* w_pre  = (const float*)d_in[1];
  const float* w_a1   = (const float*)d_in[2];
  const float* b_a1   = (const float*)d_in[3];
  const float* w_a2   = (const float*)d_in[4];
  const float* b_a2   = (const float*)d_in[5];
  const float* ghost  = (const float*)d_in[6];
  const float* w_post = (const float*)d_in[7];
  const float* b_post = (const float*)d_in[8];
  float* out = (float*)d_out;

  // workspace layout (all chunks 256B-aligned by construction)
  char* w = (char*)d_ws;
  size_t off = 0;
  auto take = [&](size_t bytes) { void* p = w + off; off += (bytes + 255) & ~(size_t)255; return p; };
  float*          qkv    = (float*)         take((size_t)PIX * NQKV * 4);      // fp32
  float*          had    = (float*)         take((size_t)PIX * DIMQK * 4);     // fp32
  float*          attn   = (float*)         take((size_t)PIX * (G * 25) * 4);  // fp32
  unsigned short* xb     = (unsigned short*)take((size_t)PIX * DIM * 2);       // bf16
  unsigned short* hb     = (unsigned short*)take((size_t)PIX * DIMQK * 2);     // bf16
  unsigned short* elsab  = (unsigned short*)take((size_t)PIX * DIMV * 2);      // bf16
  unsigned short* wpreb  = (unsigned short*)take((size_t)NQKV * DIM * 2);      // bf16
  unsigned short* wa2b   = (unsigned short*)take((size_t)384 * DIMQK * 2);     // bf16, padded 300->384
  unsigned short* wpostb = (unsigned short*)take((size_t)DIM * DIMV * 2);      // bf16

  dim3 blk(256);

  // 0) one-time fp32 -> bf16 conversions (x + the three weight matrices)
  {
    int n = PIX * DIM;
    f32_to_bf16_pad<<<dim3(n / 4 / 256), blk, 0, stream>>>(x, xb, n, n);
    n = NQKV * DIM;
    f32_to_bf16_pad<<<dim3(n / 4 / 256), blk, 0, stream>>>(w_pre, wpreb, n, n);
    int npad = 384 * DIMQK;
    f32_to_bf16_pad<<<dim3(npad / 4 / 256), blk, 0, stream>>>(w_a2, wa2b, 300 * DIMQK, npad);
    n = DIM * DIMV;
    f32_to_bf16_pad<<<dim3(n / 4 / 256), blk, 0, stream>>>(w_post, wpostb, n, n);
  }

  // 1) qkv[M,896] = x @ w_pre^T
  gemm_bf16_wmma<<<dim3(PIX / BM, NQKV / BN), blk, 0, stream>>>(
      xb, wpreb, nullptr, qkv, PIX, NQKV, DIM);
  // 2) had = q * k * scale
  had_kernel<<<dim3(PIX * (DIMQK / 4) / 256), blk, 0, stream>>>(qkv, had);
  // 3) grouped 5x5 conv + GELU -> bf16
  conv_gelu_kernel<<<dim3(PIX / 256, DIMQK / GW), blk, 0, stream>>>(
      had, w_a1, b_a1, hb);
  // 4) attn logits[M,300] = h @ w_attn2^T + b   (B padded to 384 rows)
  gemm_bf16_wmma<<<dim3(PIX / BM, 384 / BN), blk, 0, stream>>>(
      hb, wa2b, b_a2, attn, PIX, G * 25, DIMQK);
  // 5) softmax over the 25 taps, per head (in place)
  softmax25_kernel<<<dim3((PIX * G + 255) / 256), blk, 0, stream>>>(attn);
  // 6) ELSA windowed aggregation -> bf16
  elsa_kernel<<<dim3(PIX / 256, G), blk, 0, stream>>>(qkv, attn, ghost, elsab);
  // 7) out[M,384] = elsa @ w_post^T + b_post
  gemm_bf16_wmma<<<dim3(PIX / BM, DIM / BN), blk, 0, stream>>>(
      elsab, wpostb, b_post, out, PIX, DIM, DIMV);
}
